// SwitchableAttention_8658654069000
// MI455X (gfx1250) — matrix operand order repaired and checked by reading, not compiled, not run on hardware
//
#include <hip/hip_runtime.h>
#include <hip/hip_bf16.h>

typedef unsigned int   u32;
typedef unsigned short u16;

typedef __bf16 v16bf __attribute__((ext_vector_type(16)));
typedef float  v8f   __attribute__((ext_vector_type(8)));

union Frag {
    v16bf v;
    uint4 q[2];
};

// Two f32 -> packed bf16 (round-to-nearest-even) in one VALU op.
__device__ __forceinline__ u32 cvt_pk_bf16(float lo, float hi) {
    u32 r;
    asm("v_cvt_pk_bf16_f32 %0, %1, %2" : "=v"(r) : "v"(lo), "v"(hi));
    return r;
}

__device__ __forceinline__ uint2 pack4_bf16(float4 f) {
    uint2 p;
    p.x = cvt_pk_bf16(f.x, f.y);
    p.y = cvt_pk_bf16(f.z, f.w);
    return p;
}

__device__ __forceinline__ float sigmoidf(float x) {
    return 1.0f / (1.0f + __expf(-x));
}

// ---------------------------------------------------------------------------
// Kernel 1: per-batch Qt, u = Wk^T Qt, scores -> sigmoid -> attn, hot.
// scores[b,k] = (Wk^T Qt[b]) . K[b,k] + Qt[b].bk  -- Kt never materialized.
// ---------------------------------------------------------------------------
__global__ __launch_bounds__(256)
void qk_hot_kernel(const float* __restrict__ Q, const float* __restrict__ K,
                   const float* __restrict__ Wq, const float* __restrict__ bq,
                   const float* __restrict__ Wk, const float* __restrict__ bk,
                   const float* __restrict__ Wsw, const float* __restrict__ bsw,
                   float* __restrict__ attn_out,   // (512,128)
                   float* __restrict__ hot_out)    // (512,)
{
    __shared__ float Qs[768];
    __shared__ float Qt[64];
    __shared__ float us[768];
    __shared__ float attnS[128];
    __shared__ float cS;

    const int b    = blockIdx.x;
    const int tid  = threadIdx.x;
    const int wave = tid >> 5;
    const int lane = tid & 31;

    for (int i = tid; i < 768; i += 256) Qs[i] = Q[(size_t)b * 768 + i];
    __syncthreads();

    for (int h = wave; h < 58; h += 8) {
        float s = 0.f;
        const float* wr = Wq + h * 768;
        for (int d = lane; d < 768; d += 32) s += Qs[d] * wr[d];
        for (int o = 16; o > 0; o >>= 1) s += __shfl_xor(s, o, 32);
        if (lane == 0) Qt[h] = s + bq[h];
    }
    __syncthreads();

    for (int d = tid; d < 768; d += 256) {
        float s = 0.f;
        for (int h = 0; h < 58; ++h) s += Qt[h] * Wk[h * 768 + d];
        us[d] = s;
    }
    if (tid == 0) {
        float s = 0.f;
        for (int h = 0; h < 58; ++h) s += Qt[h] * bk[h];
        cS = s;
    }
    __syncthreads();
    const float cc = cS;

    const float scale = 0.0883883476483184f; // 1/sqrt(128)
    for (int k = wave; k < 128; k += 8) {
        float s = 0.f;
        const float* Kr = K + ((size_t)b * 128 + k) * 768;
        for (int d = lane; d < 768; d += 32) s += Kr[d] * us[d];
        for (int o = 16; o > 0; o >>= 1) s += __shfl_xor(s, o, 32);
        if (lane == 0) {
            float a = sigmoidf((s + cc) * scale);
            attnS[k] = a;
            attn_out[(size_t)b * 128 + k] = a;
        }
    }
    __syncthreads();

    if (wave == 0) {
        float s = 0.f;
        for (int k = lane; k < 128; k += 32) s += attnS[k] * Wsw[k];
        for (int o = 16; o > 0; o >>= 1) s += __shfl_xor(s, o, 32);
        if (lane == 0) hot_out[b] = sigmoidf(s + bsw[0]);
    }
}

// ---------------------------------------------------------------------------
// Kernel 2: context = rowscale(m) * (Vflat @ Wv^T + bv)
//   M = 133632 (= 1044*128), N = 512 (= 4*128), K = 768 (= 24*32)
//   BF16 WMMA 16x16x32, f32 acc. Block tile 128x128, wave tile 32x64.
//   Double-buffered LDS; sched_barrier-pinned phases:
//     [global prefetch issue] -> [ds loads + 8 WMMAs] -> [cvt + ds store]
// ---------------------------------------------------------------------------
__global__ __launch_bounds__(256)
void ctx_gemm_kernel(const float* __restrict__ V, const float* __restrict__ Wv,
                     const float* __restrict__ bv, const float* __restrict__ hot,
                     float* __restrict__ out)
{
    __shared__ alignas(16) u16 Vt[2][128][32];  // A tiles, bf16, [m][k]
    __shared__ alignas(16) u16 Wt[2][128][32];  // B tiles, bf16, [n][k]

    const int tid  = threadIdx.x;
    const int wave = tid >> 5;
    const int lane = tid & 31;
    const int lg   = lane >> 4;      // lane group 0/1
    const int l15  = lane & 15;

    const int m0 = blockIdx.y * 128;
    const int n0 = blockIdx.x * 128;

    // wave tile: 32 (M) x 64 (N)
    const int msub = (wave & 3) * 32;
    const int nsub = (wave >> 2) * 64;

    const int kbaseA = lg * 8;       // A: K chunks [kb,kb+8) and [kb+16,kb+24)
    const int kbaseB = lg * 8;       // B: K chunks [kb,kb+8) and [kb+16,kb+24)

    // staging geometry: each thread owns column group c4 in 4 rows r0+32i
    const int r0 = tid >> 3;
    const int c4 = (tid & 7) * 4;
    const float* Vp = V  + (size_t)(m0 + r0) * 768 + c4;
    const float* Wp = Wv + (size_t)(n0 + r0) * 768 + c4;

    // preload bias for the 4 N-subtiles this wave owns
    float bias[4];
    #pragma unroll
    for (int j = 0; j < 4; ++j) bias[j] = bv[n0 + nsub + j * 16 + l15];

    v8f c[2][4] = {};
    float4 va[4], wa[4];

    // prefetch + stage tile 0
    #pragma unroll
    for (int i = 0; i < 4; ++i) {
        va[i] = *(const float4*)(Vp + (size_t)i * 32 * 768);
        wa[i] = *(const float4*)(Wp + (size_t)i * 32 * 768);
    }
    #pragma unroll
    for (int i = 0; i < 4; ++i) {
        *(uint2*)&Vt[0][r0 + 32 * i][c4] = pack4_bf16(va[i]);
        *(uint2*)&Wt[0][r0 + 32 * i][c4] = pack4_bf16(wa[i]);
    }
    __syncthreads();

    for (int it = 0; it < 24; ++it) {
        const int cur = it & 1;
        const int nxt = cur ^ 1;

        // ---- phase 1: issue next tile's global loads
        if (it < 23) {
            const int kk = (it + 1) * 32;
            #pragma unroll
            for (int i = 0; i < 4; ++i) {
                va[i] = *(const float4*)(Vp + kk + (size_t)i * 32 * 768);
                wa[i] = *(const float4*)(Wp + kk + (size_t)i * 32 * 768);
            }
        }
        __builtin_amdgcn_sched_barrier(0);   // keep load issue above compute

        // ---- phase 2: ds loads (distinct regs) + 8 WMMAs
        Frag a[2], bfr[4];
        #pragma unroll
        for (int ai = 0; ai < 2; ++ai) {
            const int arow = msub + ai * 16 + l15;
            a[ai].q[0] = *(const uint4*)&Vt[cur][arow][kbaseA];
            a[ai].q[1] = *(const uint4*)&Vt[cur][arow][kbaseA + 16];
        }
        #pragma unroll
        for (int j = 0; j < 4; ++j) {
            const int brow = nsub + j * 16 + l15;
            bfr[j].q[0] = *(const uint4*)&Wt[cur][brow][kbaseB];
            bfr[j].q[1] = *(const uint4*)&Wt[cur][brow][kbaseB + 16];
        }
        #pragma unroll
        for (int j = 0; j < 4; ++j) {
            #pragma unroll
            for (int ai = 0; ai < 2; ++ai) {
                c[ai][j] = __builtin_amdgcn_wmma_f32_16x16x32_bf16(
                    false, a[ai].v, false, bfr[j].v, (short)0, c[ai][j],
                    false, false);
            }
        }
        __builtin_amdgcn_sched_barrier(0);   // keep stores below compute

        // ---- phase 3: convert + store next tile into the other buffer
        if (it < 23) {
            #pragma unroll
            for (int i = 0; i < 4; ++i) {
                *(uint2*)&Vt[nxt][r0 + 32 * i][c4] = pack4_bf16(va[i]);
                *(uint2*)&Wt[nxt][r0 + 32 * i][c4] = pack4_bf16(wa[i]);
            }
        }
        __syncthreads();
    }

    // epilogue: bias + switchable scale, non-temporal streaming stores
    #pragma unroll
    for (int ai = 0; ai < 2; ++ai) {
        #pragma unroll
        for (int r = 0; r < 8; ++r) {
            const int m  = m0 + msub + ai * 16 + r + lg * 8; // C: VGPR r -> M=r(+8)
            const int bb = m / 261;
            const int s  = m - bb * 261;
            const float h = hot[bb];
            const float scl = (s < 197) ? h : (1.0f - h);
            #pragma unroll
            for (int j = 0; j < 4; ++j) {
                const int n = n0 + nsub + j * 16 + l15;
                __builtin_nontemporal_store(scl * (c[ai][j][r] + bias[j]),
                                            out + (size_t)m * 512 + n);
            }
        }
    }
}

extern "C" void kernel_launch(void* const* d_in, const int* in_sizes, int n_in,
                              void* d_out, int out_size, void* d_ws, size_t ws_size,
                              hipStream_t stream) {
    const float* Q   = (const float*)d_in[0];
    const float* K   = (const float*)d_in[1];
    const float* V   = (const float*)d_in[2];
    const float* Wq  = (const float*)d_in[3];
    const float* bq  = (const float*)d_in[4];
    const float* Wk  = (const float*)d_in[5];
    const float* bk  = (const float*)d_in[6];
    const float* Wsw = (const float*)d_in[7];
    const float* bsw = (const float*)d_in[8];
    const float* Wv  = (const float*)d_in[9];
    const float* bv  = (const float*)d_in[10];

    float* out = (float*)d_out;
    float* hot = (float*)d_ws;                  // 512 floats of scratch

    const size_t CTX = (size_t)512 * 261 * 512; // context elements; attn follows

    qk_hot_kernel<<<512, 256, 0, stream>>>(Q, K, Wq, bq, Wk, bk, Wsw, bsw,
                                           out + CTX, hot);

    ctx_gemm_kernel<<<dim3(4, 1044), 256, 0, stream>>>(V, Wv, bv, hot, out);
}
